// GatedAxialAttentionUnit_68143951119084
// MI455X (gfx1250) — matrix operand order repaired and checked by reading, not compiled, not run on hardware
//
// Gated axial attention for MI455X (gfx1250), fused single kernel.
// One workgroup (8 wave32 waves) per axial slab bs in [0,1024); wave w owns
// head w. All GEMMs via v_wmma_f32_16x16x32_f16. Weights pre-converted to f16
// once into d_ws by a small kernel; A/B fragments software-pipelined.
#include <hip/hip_runtime.h>

typedef __attribute__((ext_vector_type(16))) _Float16 v16h;
typedef __attribute__((ext_vector_type(8)))  _Float16 v8h;
typedef __attribute__((ext_vector_type(8)))  float    v8f;

#define T_DIM   128
#define C_DIM   256
#define S_DIM   1024
#define N_HEADS 8
#define H_DIM   32
#define SCALE_Q 0.17677669529663687f   // 1/sqrt(32)
#define W_ELEMS 65536                  // 256*256 per weight matrix

// ---------------- WMMA helpers ----------------
__device__ __forceinline__ v8f wmma16(v16h a, v16h b, v8f c) {
  return __builtin_amdgcn_wmma_f32_16x16x32_f16(false, a, false, b, (short)0, c,
                                                false, false);
}

// A-operand fragment (16x32 f16): lane = row (l&15); halves {k0..k0+7, k0+16..k0+23},
// k0 = kc + 8*(lane>=16).  (ISA 7.12.2, 16-bit A matrix)
__device__ __forceinline__ v16h frag_a(const unsigned short* base, int row,
                                       int stride, int kc, int hi) {
  int off = row * stride + kc + hi * 8;
  union { v16h v; v8h h[2]; } u;
  u.h[0] = *reinterpret_cast<const v8h*>(base + off);
  u.h[1] = *reinterpret_cast<const v8h*>(base + off + 16);
  return u.v;
}

// B-operand fragment (32x16 f16): lane holds B^T row (l&15); halves {k0..k0+7,
// k0+16..k0+23}, k0 = kc + 8*(lane>=16). Works for LDS or global f16 storage.
__device__ __forceinline__ v16h frag_b(const unsigned short* base, int row,
                                       int stride, int kc, int hi) {
  int off = row * stride + kc + hi * 8;
  union { v16h v; v8h h[2]; } u;
  u.h[0] = *reinterpret_cast<const v8h*>(base + off);
  u.h[1] = *reinterpret_cast<const v8h*>(base + off + 16);
  return u.v;
}

__device__ __forceinline__ unsigned short f2h(float x) {
  union { _Float16 h; unsigned short u; } t; t.h = (_Float16)x; return t.u;
}
__device__ __forceinline__ unsigned int packh2(float a, float b) {
  union { _Float16 h[2]; unsigned int u; } t;
  t.h[0] = (_Float16)a; t.h[1] = (_Float16)b; return t.u;
}
__device__ __forceinline__ float unpackh2(unsigned int u, int i) {
  union { unsigned int u; _Float16 h[2]; } t; t.u = u; return (float)t.h[i];
}

// [128x256(f16 LDS)] x [256 -> 2 col tiles of f16 weights]: wave w computes
// output cols 32w..32w+31 for all 128 rows. Weight fragments reused across the
// 8 M-tiles; A and B fragment fetches pipelined one step ahead of their WMMAs.
__device__ __forceinline__ void gemm_128x256(const unsigned short* __restrict__ Wh,
                                             const unsigned short* sa,
                                             int w, int lr, int hi,
                                             v8f (&acc)[8][2]) {
  v8f zero = {};
#pragma unroll
  for (int mt = 0; mt < 8; ++mt) { acc[mt][0] = zero; acc[mt][1] = zero; }
  const int n0 = w * 32 + lr, n1 = n0 + 16;
  v16h b0 = frag_b(Wh, n0, 256, 0, hi);
  v16h b1 = frag_b(Wh, n1, 256, 0, hi);
#pragma unroll
  for (int ks = 0; ks < 8; ++ks) {
    int kc = ks * 32;
    v16h nb0 = b0, nb1 = b1;
    if (ks < 7) {                      // prefetch next K-step's weights
      nb0 = frag_b(Wh, n0, 256, kc + 32, hi);
      nb1 = frag_b(Wh, n1, 256, kc + 32, hi);
    }
    v16h a = frag_a(sa, lr, 256, kc, hi);
#pragma unroll
    for (int mt = 0; mt < 8; ++mt) {
      v16h an = a;
      if (mt < 7) an = frag_a(sa, (mt + 1) * 16 + lr, 256, kc, hi);
      acc[mt][0] = wmma16(a, b0, acc[mt][0]);
      acc[mt][1] = wmma16(a, b1, acc[mt][1]);
      a = an;
    }
    b0 = nb0; b1 = nb1;
  }
}

// ---------------- weight pre-conversion (runs once per launch) --------------
__global__ void __launch_bounds__(256)
cvt_w_kernel(const float* __restrict__ Wq, const float* __restrict__ Wk,
             const float* __restrict__ Wv, const float* __restrict__ Wg,
             const float* __restrict__ Wo, unsigned short* __restrict__ wh) {
  int t = blockIdx.x * 256 + threadIdx.x;   // 81920 threads, 4 floats each
  int m = t >> 14;                          // 16384 threads per matrix
  int off = (t & 16383) * 4;
  const float* src = (m == 0) ? Wq : (m == 1) ? Wk : (m == 2) ? Wv
                   : (m == 3) ? Wg : Wo;
  float4 f = *reinterpret_cast<const float4*>(src + off);
  union { _Float16 h[4]; unsigned long long u; } p;
  p.h[0] = (_Float16)f.x; p.h[1] = (_Float16)f.y;
  p.h[2] = (_Float16)f.z; p.h[3] = (_Float16)f.w;
  *reinterpret_cast<unsigned long long*>(wh + m * W_ELEMS + off) = p.u;
}

// LDS (f16 halves): a/z slab 128x256 = 32768, then per-wave private:
//   q 128x32 = 4096 | k 128x32 = 4096 | vT 32x128 = 4096 | p 16x128 = 2048
#define PW_SZ   14336
#define LDS_TOT (32768 + 8 * PW_SZ)   // 147456 halves = 288 KB (<= 320 KB/WGP)

__global__ void __launch_bounds__(256)
gaau_kernel(const float* __restrict__ x, const float* __restrict__ bias,
            const unsigned short* __restrict__ wh,
            const float* __restrict__ bg, const float* __restrict__ bo,
            float* __restrict__ out) {
  __shared__ __align__(16) unsigned short smem[LDS_TOT];

  const unsigned short* Wqh = wh;
  const unsigned short* Wkh = wh + 1 * W_ELEMS;
  const unsigned short* Wvh = wh + 2 * W_ELEMS;
  const unsigned short* Wgh = wh + 3 * W_ELEMS;
  const unsigned short* Woh = wh + 4 * W_ELEMS;

  const int bs   = blockIdx.x;        // axial position 0..1023
  const int tid  = threadIdx.x;
  const int w    = tid >> 5;          // wave id == head id (wave32)
  const int lane = tid & 31;
  const int lr   = lane & 15;
  const int hi   = lane >> 4;

  unsigned short* qb  = smem + 32768 + w * PW_SZ;
  unsigned short* kb  = qb + 4096;
  unsigned short* vtb = qb + 8192;
  unsigned short* pb  = qb + 12288;

  // ---- Phase 0: stage a[bs] = x[0, t, bs, :] into LDS as f16 (row-major 128x256)
  {
    int row = tid >> 1;
    int cb  = (tid & 1) * 128;
    const float* src = x + (size_t)row * (S_DIM * C_DIM) + (size_t)bs * C_DIM + cb;
    unsigned short* dst = smem + row * 256 + cb;
#pragma unroll
    for (int i = 0; i < 32; ++i) {
      float4 f = reinterpret_cast<const float4*>(src)[i];
      union { _Float16 h[4]; unsigned long long u; } t;
      t.h[0] = (_Float16)f.x; t.h[1] = (_Float16)f.y;
      t.h[2] = (_Float16)f.z; t.h[3] = (_Float16)f.w;
      *reinterpret_cast<unsigned long long*>(dst + i * 4) = t.u;
    }
  }
  __syncthreads();

  // ---- Phase 1: projections (K, V, Q, G) for this wave's head columns
  v8f acc[8][2];
  unsigned int gpk[8][2][4];   // gate, packed f16 pairs, kept in registers

  // K -> kb row-major [s][d]
  gemm_128x256(Wkh, smem, w, lr, hi, acc);
#pragma unroll
  for (int mt = 0; mt < 8; ++mt)
#pragma unroll
    for (int ntp = 0; ntp < 2; ++ntp)
#pragma unroll
      for (int r = 0; r < 8; ++r)
        kb[(mt * 16 + hi * 8 + r) * 32 + ntp * 16 + lr] = f2h(acc[mt][ntp][r]);

  // V -> vtb transposed [d][s]; each lane's 8 rows are contiguous s -> b128 store
  gemm_128x256(Wvh, smem, w, lr, hi, acc);
#pragma unroll
  for (int mt = 0; mt < 8; ++mt)
#pragma unroll
    for (int ntp = 0; ntp < 2; ++ntp) {
      union { _Float16 h[8]; uint4 u; } t;
#pragma unroll
      for (int r = 0; r < 8; ++r) t.h[r] = (_Float16)acc[mt][ntp][r];
      *reinterpret_cast<uint4*>(vtb + (ntp * 16 + lr) * 128 + mt * 16 + hi * 8) = t.u;
    }

  // Q (pre-scaled by 1/sqrt(d)) -> qb row-major [t][d]
  gemm_128x256(Wqh, smem, w, lr, hi, acc);
#pragma unroll
  for (int mt = 0; mt < 8; ++mt)
#pragma unroll
    for (int ntp = 0; ntp < 2; ++ntp)
#pragma unroll
      for (int r = 0; r < 8; ++r)
        qb[(mt * 16 + hi * 8 + r) * 32 + ntp * 16 + lr] =
            f2h(acc[mt][ntp][r] * SCALE_Q);

  // G = sigmoid(a@Wg^T + bg) -> packed registers
  gemm_128x256(Wgh, smem, w, lr, hi, acc);
#pragma unroll
  for (int mt = 0; mt < 8; ++mt)
#pragma unroll
    for (int ntp = 0; ntp < 2; ++ntp) {
      float bgc = bg[w * 32 + ntp * 16 + lr];
#pragma unroll
      for (int j = 0; j < 4; ++j) {
        float g0 = 1.0f / (1.0f + __expf(-(acc[mt][ntp][2 * j + 0] + bgc)));
        float g1 = 1.0f / (1.0f + __expf(-(acc[mt][ntp][2 * j + 1] + bgc)));
        gpk[mt][ntp][j] = packh2(g0, g1);
      }
    }

  __syncthreads();   // everyone done reading 'a'; smem slab now becomes 'z'

  // ---- Phase 2: attention for head w, one 16-row tile at a time
  const float* bptr = bias + w;   // attn_bias[0, t, s, w]
  for (int mt = 0; mt < 8; ++mt) {
    v16h qf = frag_a(qb, mt * 16 + lr, 32, 0, hi);
    v8f zero = {};
    v8f s[8];
    v16h kf = frag_b(kb, lr, 32, 0, hi);
#pragma unroll
    for (int nt = 0; nt < 8; ++nt) {
      v16h kn = kf;
      if (nt < 7) kn = frag_b(kb, (nt + 1) * 16 + lr, 32, 0, hi);
      s[nt] = wmma16(qf, kf, zero);
      kf = kn;
    }
    // + bias (L2-resident, 512 KB total)
    int tb = mt * 16 + hi * 8;
#pragma unroll
    for (int nt = 0; nt < 8; ++nt) {
      int sc = nt * 16 + lr;
#pragma unroll
      for (int r = 0; r < 8; ++r)
        s[nt][r] += bptr[((tb + r) * T_DIM + sc) * N_HEADS];
    }
    // softmax over s (row r lives in half-wave lanes; reduce with xor 1/2/4/8)
#pragma unroll
    for (int r = 0; r < 8; ++r) {
      float mx = s[0][r];
#pragma unroll
      for (int nt = 1; nt < 8; ++nt) mx = fmaxf(mx, s[nt][r]);
      mx = fmaxf(mx, __shfl_xor(mx, 1));
      mx = fmaxf(mx, __shfl_xor(mx, 2));
      mx = fmaxf(mx, __shfl_xor(mx, 4));
      mx = fmaxf(mx, __shfl_xor(mx, 8));
      float sum = 0.f;
#pragma unroll
      for (int nt = 0; nt < 8; ++nt) {
        float e = __expf(s[nt][r] - mx);
        s[nt][r] = e;
        sum += e;
      }
      sum += __shfl_xor(sum, 1);
      sum += __shfl_xor(sum, 2);
      sum += __shfl_xor(sum, 4);
      sum += __shfl_xor(sum, 8);
      float inv = 1.0f / sum;
#pragma unroll
      for (int nt = 0; nt < 8; ++nt)
        pb[(hi * 8 + r) * 128 + nt * 16 + lr] = f2h(s[nt][r] * inv);
    }
    // O = P @ V  (K = 128 over s, vT gives contiguous B fragments)
    v8f o0 = zero, o1 = zero;
    v16h v0 = frag_b(vtb, lr, 128, 0, hi);
    v16h v1 = frag_b(vtb, 16 + lr, 128, 0, hi);
#pragma unroll
    for (int ks = 0; ks < 4; ++ks) {
      v16h nv0 = v0, nv1 = v1;
      if (ks < 3) {
        nv0 = frag_b(vtb, lr, 128, ks * 32 + 32, hi);
        nv1 = frag_b(vtb, 16 + lr, 128, ks * 32 + 32, hi);
      }
      v16h pf = frag_a(pb, lr, 128, ks * 32, hi);
      o0 = wmma16(pf, v0, o0);
      o1 = wmma16(pf, v1, o1);
      v0 = nv0; v1 = nv1;
    }
    // z = gate * O -> back into the (now dead) 'a' slab
#pragma unroll
    for (int r = 0; r < 8; ++r) {
      int t_ = mt * 16 + hi * 8 + r;
      float g0 = unpackh2(gpk[mt][0][r >> 1], r & 1);
      float g1 = unpackh2(gpk[mt][1][r >> 1], r & 1);
      smem[t_ * 256 + w * 32 + lr]      = f2h(g0 * o0[r]);
      smem[t_ * 256 + w * 32 + 16 + lr] = f2h(g1 * o1[r]);
    }
  }

  __syncthreads();   // z complete across all heads

  // ---- Phase 3: output projection  out = z @ Wo^T + bo  (same GEMM shape)
  v8f acc2[8][2];
  gemm_128x256(Woh, smem, w, lr, hi, acc2);
  float bo0 = bo[w * 32 + lr];
  float bo1 = bo[w * 32 + 16 + lr];
#pragma unroll
  for (int mt = 0; mt < 8; ++mt)
#pragma unroll
    for (int r = 0; r < 8; ++r) {
      int t_ = mt * 16 + hi * 8 + r;
      size_t base = (size_t)t_ * (S_DIM * C_DIM) + (size_t)bs * C_DIM;
      out[base + w * 32 + lr]      = acc2[mt][0][r] + bo0;
      out[base + w * 32 + 16 + lr] = acc2[mt][1][r] + bo1;
    }
}

extern "C" void kernel_launch(void* const* d_in, const int* in_sizes, int n_in,
                              void* d_out, int out_size, void* d_ws, size_t ws_size,
                              hipStream_t stream) {
  const float* x  = (const float*)d_in[0];
  const float* ab = (const float*)d_in[1];
  const float* Wq = (const float*)d_in[2];
  const float* Wk = (const float*)d_in[3];
  const float* Wv = (const float*)d_in[4];
  const float* Wg = (const float*)d_in[5];
  const float* bg = (const float*)d_in[6];
  const float* Wo = (const float*)d_in[7];
  const float* bo = (const float*)d_in[8];
  float* out = (float*)d_out;
  unsigned short* wh = (unsigned short*)d_ws;   // 5 * 65536 f16 = 640 KB

  cvt_w_kernel<<<dim3(320), dim3(256), 0, stream>>>(Wq, Wk, Wv, Wg, Wo, wh);
  gaau_kernel<<<dim3(S_DIM), dim3(256), 0, stream>>>(x, ab, wh, bg, bo, out);
}
